// TimeLSTM_3770981286526
// MI455X (gfx1250) — matrix operand order repaired and checked by reading, not compiled, not run on hardware
//
#include <hip/hip_runtime.h>

// TimeLSTM for MI455X (gfx1250, wave32, WMMA).
// Phase 1: convert inputs/weights f32->bf16.
// Phase 2: u = x @ U^T + b  via double-buffered async-LDS bf16 WMMA GEMM.
// Phase 3: persistent scan kernel, 40 blocks; each block keeps its 128-column
//          weight slice (258KB bf16) resident in LDS for all 512 steps;
//          steps separated by a device-wide sense barrier.
// Streaming buffers (u, outputs) use non-temporal cache hints to protect L2.

typedef __attribute__((ext_vector_type(16))) __bf16 v16bf;
typedef __attribute__((ext_vector_type(8)))  __bf16 v8bf;
typedef __attribute__((ext_vector_type(8)))  float  v8f;
typedef __attribute__((ext_vector_type(4)))  int    v4i_;

#define B_   64
#define S_   512
#define D_   1024
#define H_   1024
#define G4_  4096
#define ROWS_ (S_ * B_)       // 32768
#define NBLK_SCAN 40
#define TPB 256

// ---- gfx1250 async global->LDS path (ASYNCcnt), with safe fallback ----
#if defined(__HIP_DEVICE_COMPILE__) && __has_builtin(__builtin_amdgcn_global_load_async_to_lds_b128)
#define ASYNC_LDS 1
#else
#define ASYNC_LDS 0
#endif

__device__ __forceinline__ void cp16_async(void* lds, const void* g) {
#if ASYNC_LDS
  // Probed signature: (v4i addrspace(1)*, v4i addrspace(3)*, imm offset, imm cpol)
  __builtin_amdgcn_global_load_async_to_lds_b128(
      (__attribute__((address_space(1))) v4i_*)const_cast<void*>(g),
      (__attribute__((address_space(3))) v4i_*)lds, 0, 0);
#else
  *(v8bf*)lds = *(const v8bf*)g;
#endif
}

__device__ __forceinline__ void async_wait0() {
#if ASYNC_LDS
#if __has_builtin(__builtin_amdgcn_s_wait_asynccnt)
  __builtin_amdgcn_s_wait_asynccnt(0);
#else
  asm volatile("s_wait_asynccnt 0x0" ::: "memory");
#endif
#endif
}

__device__ __forceinline__ float sigmoidf_(float x) {
  return 1.0f / (1.0f + __expf(-x));
}

__device__ __forceinline__ v8f vzero8() {
  v8f z;
#pragma unroll
  for (int t = 0; t < 8; ++t) z[t] = 0.0f;
  return z;
}

__device__ __forceinline__ v16bf pack16(v8bf lo, v8bf hi) {
  v16bf r;
#pragma unroll
  for (int t = 0; t < 8; ++t) { r[t] = lo[t]; r[t + 8] = hi[t]; }
  return r;
}

// ---------------- init: zero recurrent state + barrier ----------------
__global__ void k_init(float* cf, __bf16* hbf, __bf16* cbf, int* bar) {
  int i = blockIdx.x * blockDim.x + threadIdx.x;
  for (int j = i; j < B_ * H_; j += gridDim.x * blockDim.x) {
    cf[j] = 0.0f;
    hbf[j] = (__bf16)0.0f;
    cbf[j] = (__bf16)0.0f;
  }
  if (i < 2) bar[i] = 0;
}

// ---------------- converts ----------------
// inputs [B,S,D] f32 -> xb [(s*B+b), D] bf16  (row = s*B+b matches u layout)
__global__ void k_cvt_x(const float* __restrict__ x, __bf16* __restrict__ xb) {
  const size_t n = (size_t)B_ * S_ * D_;
  for (size_t i = (size_t)blockIdx.x * blockDim.x + threadIdx.x; i < n;
       i += (size_t)gridDim.x * blockDim.x) {
    size_t b = i / ((size_t)S_ * D_);
    size_t r = i % ((size_t)S_ * D_);
    size_t s = r / D_, d = r % D_;
    xb[(s * B_ + b) * D_ + d] = (__bf16)x[i];
  }
}

__global__ void k_cvt(const float* __restrict__ src, __bf16* __restrict__ dst, int n) {
  for (int i = blockIdx.x * blockDim.x + threadIdx.x; i < n;
       i += gridDim.x * blockDim.x)
    dst[i] = (__bf16)src[i];
}

// ---------------- u = xb @ U^T + bias  (M=32768, N=4096, K=1024) ----------------
// Block tile 128x128, 8 waves as 4(M)x2(N); wave tile 32x64 -> 2x4 wmma accums.
// Double-buffered LDS staging via GLOBAL_LOAD_ASYNC_TO_LDS_B128.
#define LDA_S 40  // padded LDS row stride (bf16): 80B -> 16B aligned, bank-spread
__global__ void __launch_bounds__(TPB) k_gemm_u(
    const __bf16* __restrict__ xb, const __bf16* __restrict__ ubf,
    const float* __restrict__ ubias, float* __restrict__ u) {
  __shared__ __bf16 As[2][128 * LDA_S];
  __shared__ __bf16 Bs[2][128 * LDA_S];
  const int tid = threadIdx.x, lane = tid & 31, wave = tid >> 5;
  const int wm = wave & 3;            // M sub-block: wm*32
  const int wn = wave >> 2;           // N sub-block: wn*64
  const size_t m0 = (size_t)blockIdx.y * 128;
  const int n0 = blockIdx.x * 128;

  v8f acc[2][4];
#pragma unroll
  for (int mi = 0; mi < 2; ++mi)
#pragma unroll
    for (int ni = 0; ni < 4; ++ni) acc[mi][ni] = vzero8();

  const int rr = tid >> 2;            // 0..63
  const int kc = (tid & 3) << 3;      // 0,8,16,24
  const int m_ = lane & 15;
  const int ka = (lane >> 4) << 3;    // A frag K sub-chunk: 0 / 8
  const int kb = (lane >> 4) << 4;    // unused; B frags use ka (0 / 8)

  auto stage = [&](int k0, int buf) {
    cp16_async(&As[buf][rr * LDA_S + kc],        &xb[(m0 + rr) * D_ + k0 + kc]);
    cp16_async(&As[buf][(rr + 64) * LDA_S + kc], &xb[(m0 + rr + 64) * D_ + k0 + kc]);
    cp16_async(&Bs[buf][rr * LDA_S + kc],        &ubf[(size_t)(n0 + rr) * D_ + k0 + kc]);
    cp16_async(&Bs[buf][(rr + 64) * LDA_S + kc], &ubf[(size_t)(n0 + rr + 64) * D_ + k0 + kc]);
  };

  stage(0, 0);
  async_wait0();
  __syncthreads();

  for (int k0 = 0; k0 < D_; k0 += 32) {
    const int cur = (k0 >> 5) & 1;
    if (k0 + 32 < D_) stage(k0 + 32, cur ^ 1);  // overlap with this chunk's WMMAs

    v16bf af[2], bfr[4];
#pragma unroll
    for (int mi = 0; mi < 2; ++mi) {
      const __bf16* pa = &As[cur][(wm * 32 + mi * 16 + m_) * LDA_S];
      af[mi] = pack16(*(const v8bf*)(pa + ka), *(const v8bf*)(pa + 16 + ka));
    }
#pragma unroll
    for (int ni = 0; ni < 4; ++ni) {
      const __bf16* pb = &Bs[cur][(wn * 64 + ni * 16 + m_) * LDA_S + ka];
      bfr[ni] = pack16(*(const v8bf*)pb, *(const v8bf*)(pb + 16));
    }
#pragma unroll
    for (int mi = 0; mi < 2; ++mi)
#pragma unroll
      for (int ni = 0; ni < 4; ++ni)
        acc[mi][ni] = __builtin_amdgcn_wmma_f32_16x16x32_bf16(
            false, af[mi], false, bfr[ni], (short)0, acc[mi][ni], false, false);

    async_wait0();     // next buffer fully in LDS (my wave's ASYNCcnt)
    __syncthreads();   // all waves' stages done; reads of cur done
  }

  const size_t rowb = m0 + wm * 32 + ((lane >> 4) << 3);
#pragma unroll
  for (int mi = 0; mi < 2; ++mi)
#pragma unroll
    for (int ni = 0; ni < 4; ++ni) {
      const int col = n0 + wn * 64 + ni * 16 + m_;
      const float bias = ubias[col];
#pragma unroll
      for (int r = 0; r < 8; ++r)
        __builtin_nontemporal_store(acc[mi][ni][r] + bias,
                                    &u[(rowb + mi * 16 + r) * G4_ + col]);
    }
}

// ---------------- device-wide sense barrier ----------------
__device__ __forceinline__ void gbar(int* cnt, int* gen, int nb) {
  __threadfence();
  __syncthreads();
  if (threadIdx.x == 0) {
    volatile int* vgen = (volatile int*)gen;
    int g = *vgen;
    if (atomicAdd(cnt, 1) == nb - 1) {
      *cnt = 0;
      __threadfence();
      atomicAdd(gen, 1);
    } else {
      while (*vgen == g) __builtin_amdgcn_s_sleep(1);
    }
  }
  __syncthreads();
  __threadfence();
}

// ---------------- persistent scan ----------------
// blk 0..31: gates tile (A=h_bf, B=W_all slice) -> gates[64,4096] raw
// blk 32..39: decay tile (A=c_bf, B=W_d  slice) -> cs1[64,1024] raw
// Weight slice (128 cols x 1024 K bf16, row stride 1032) lives in LDS all run.
#define LDW_S 1032  // 2064B rows: 16B aligned, banks rotate by 4 dwords/row
__global__ void __launch_bounds__(TPB, 1) k_scan(
    const __bf16* __restrict__ Wbf, const __bf16* __restrict__ Wdbf,
    const float* __restrict__ Wb, const float* __restrict__ Wdb,
    const float* __restrict__ u, const float* __restrict__ ts,
    float* __restrict__ gates, float* __restrict__ cs1,
    float* __restrict__ cf,
    __bf16* __restrict__ hbf, __bf16* __restrict__ cbf,
    float* __restrict__ outO, float* __restrict__ outH, float* __restrict__ outC,
    int* bar) {
  extern __shared__ __bf16 smem[];  // 128 * LDW_S bf16 = 258KB
  const int tid = threadIdx.x, lane = tid & 31, wave = tid >> 5;
  const int blk = blockIdx.x;
  const bool is_gate = (blk < 32);

  // one-time weight slice load into LDS (async DMA, resident for 512 steps)
  const __bf16* wsrc = is_gate ? (Wbf + (size_t)blk * 128 * H_)
                               : (Wdbf + (size_t)(blk - 32) * 128 * H_);
  for (int idx = tid; idx < 128 * 128; idx += TPB) {
    int r = idx >> 7;
    int kk = (idx & 127) << 3;
    cp16_async(&smem[r * LDW_S + kk], &wsrc[(size_t)r * H_ + kk]);
  }
  async_wait0();
  __syncthreads();

  const __bf16* Abf = is_gate ? hbf : cbf;
  float* obuf = is_gate ? (gates + blk * 128) : (cs1 + (blk - 32) * 128);
  const int ldo = is_gate ? G4_ : H_;

  const int wm = wave & 1;          // M: 2 waves * 32 rows
  const int wn = wave >> 1;         // N: 4 waves * 32 cols
  const int m_ = lane & 15;
  const int ka = (lane >> 4) << 3;
  const int kb = (lane >> 4) << 4;

  for (int s = 0; s < S_; ++s) {
    // ---- GEMM 64x128, K=1024 ----
    v8f acc[2][2];
#pragma unroll
    for (int mi = 0; mi < 2; ++mi)
#pragma unroll
      for (int ni = 0; ni < 2; ++ni) acc[mi][ni] = vzero8();

    for (int k0 = 0; k0 < H_; k0 += 32) {
      v16bf af[2], bfr[2];
#pragma unroll
      for (int mi = 0; mi < 2; ++mi) {
        const __bf16* pa = Abf + (size_t)(wm * 32 + mi * 16 + m_) * H_ + k0;
        af[mi] = pack16(*(const v8bf*)(pa + ka), *(const v8bf*)(pa + 16 + ka));
      }
#pragma unroll
      for (int ni = 0; ni < 2; ++ni) {
        const __bf16* pb = &smem[(wn * 32 + ni * 16 + m_) * LDW_S + k0 + ka];
        bfr[ni] = pack16(*(const v8bf*)pb, *(const v8bf*)(pb + 16));
      }
#pragma unroll
      for (int mi = 0; mi < 2; ++mi)
#pragma unroll
        for (int ni = 0; ni < 2; ++ni)
          acc[mi][ni] = __builtin_amdgcn_wmma_f32_16x16x32_bf16(
              false, af[mi], false, bfr[ni], (short)0, acc[mi][ni], false, false);
    }
    {
      const int rowb = wm * 32 + ((lane >> 4) << 3);
      const int colb = wn * 32 + m_;
#pragma unroll
      for (int mi = 0; mi < 2; ++mi)
#pragma unroll
        for (int ni = 0; ni < 2; ++ni)
#pragma unroll
          for (int r = 0; r < 8; ++r)
            obuf[(size_t)(rowb + mi * 16 + r) * ldo + colb + ni * 16] = acc[mi][ni][r];
    }
    gbar(bar, bar + 1, NBLK_SCAN);

    // ---- elementwise cell update (all 40 blocks share 64*1024 items) ----
    for (int j = blk * TPB + tid; j < B_ * H_; j += NBLK_SCAN * TPB) {
      const int b = j >> 10;
      const int g = j & (H_ - 1);
      const size_t gb = (size_t)b * G4_ + g;
      const size_t ub = ((size_t)s * B_ + b) * G4_ + g;
      const float fr = sigmoidf_(gates[gb]        + Wb[g]        + __builtin_nontemporal_load(&u[ub]));
      const float ir = sigmoidf_(gates[gb + 1024] + Wb[g + 1024] + __builtin_nontemporal_load(&u[ub + 1024]));
      const float og = sigmoidf_(gates[gb + 2048] + Wb[g + 2048] + __builtin_nontemporal_load(&u[ub + 2048]));
      const float gg = sigmoidf_(gates[gb + 3072] + Wb[g + 3072] + __builtin_nontemporal_load(&u[ub + 3072]));
      const float c1 = tanhf(cs1[(size_t)b * H_ + g] + Wdb[g]);
      const float cp = cf[j];
      const float tt = ts[(size_t)b * S_ + s];
      const float cadj = (cp - c1) + c1 * tt;
      const float cn = fr * cadj + ir * gg;
      const float hn = og * tanhf(cn);
      cf[j] = cn;
      cbf[j] = (__bf16)cn;
      hbf[j] = (__bf16)hn;
      __builtin_nontemporal_store(og, &outO[((size_t)b * S_ + s) * H_ + g]);
      if (s == S_ - 1) { outH[j] = hn; outC[j] = cn; }
    }
    gbar(bar, bar + 1, NBLK_SCAN);
  }
}

// ---------------- host ----------------
extern "C" void kernel_launch(void* const* d_in, const int* in_sizes, int n_in,
                              void* d_out, int out_size, void* d_ws, size_t ws_size,
                              hipStream_t stream) {
  const float* inputs  = (const float*)d_in[0];
  const float* tstamp  = (const float*)d_in[1];
  const float* W_all_w = (const float*)d_in[2];
  const float* W_all_b = (const float*)d_in[3];
  const float* U_all_w = (const float*)d_in[4];
  const float* U_all_b = (const float*)d_in[5];
  const float* W_d_w   = (const float*)d_in[6];
  const float* W_d_b   = (const float*)d_in[7];

  char* ws = (char*)d_ws;
  size_t off = 0;
  auto take = [&](size_t bytes) -> char* {
    char* p = ws + off;
    off = (off + bytes + 255) & ~(size_t)255;
    return p;
  };
  __bf16* xb    = (__bf16*)take((size_t)ROWS_ * D_ * 2);
  __bf16* Ubf   = (__bf16*)take((size_t)G4_ * D_ * 2);
  __bf16* Wbf   = (__bf16*)take((size_t)G4_ * H_ * 2);
  __bf16* Wdbf  = (__bf16*)take((size_t)H_ * H_ * 2);
  float*  u     = (float*)take((size_t)ROWS_ * G4_ * 4);
  float*  gates = (float*)take((size_t)B_ * G4_ * 4);
  float*  cs1   = (float*)take((size_t)B_ * H_ * 4);
  float*  cf    = (float*)take((size_t)B_ * H_ * 4);
  __bf16* hbf   = (__bf16*)take((size_t)B_ * H_ * 2);
  __bf16* cbf   = (__bf16*)take((size_t)B_ * H_ * 2);
  int*    bar   = (int*)take(256);

  float* outO = (float*)d_out;
  float* outH = outO + (size_t)B_ * S_ * H_;
  float* outC = outH + (size_t)B_ * H_;

  // idempotent config (not stream work; identical every call)
  (void)hipFuncSetAttribute((const void*)k_scan,
                            hipFuncAttributeMaxDynamicSharedMemorySize,
                            128 * LDW_S * 2);

  k_init<<<64, TPB, 0, stream>>>(cf, hbf, cbf, bar);
  k_cvt_x<<<2048, TPB, 0, stream>>>(inputs, xb);
  k_cvt<<<1024, TPB, 0, stream>>>(U_all_w, Ubf, G4_ * D_);
  k_cvt<<<1024, TPB, 0, stream>>>(W_all_w, Wbf, G4_ * H_);
  k_cvt<<<256, TPB, 0, stream>>>(W_d_w, Wdbf, H_ * H_);
  k_gemm_u<<<dim3(G4_ / 128, ROWS_ / 128), TPB, 0, stream>>>(xb, Ubf, U_all_b, u);
  k_scan<<<NBLK_SCAN, TPB, 128 * LDW_S * 2, stream>>>(
      Wbf, Wdbf, W_all_b, W_d_b, u, tstamp,
      gates, cs1, cf, hbf, cbf, outO, outH, outC, bar);
}
